// MHMoE_9758165696686
// MI455X (gfx1250) — matrix operand order repaired and checked by reading, not compiled, not run on hardware
//
#include <hip/hip_runtime.h>

// ---------------------------------------------------------------------------
// MI455X (gfx1250) wave32 WMMA implementation of multi-head MoE forward.
// Pipeline: convert/transpose weights to bf16 once -> xh GEMM -> gating ->
// fused up/relu^2*gate/down MoE GEMMs -> output GEMM.
// All matrix math: v_wmma_f32_16x16x32_bf16 (fp32 accumulate).
// Staging global->LDS uses GLOBAL_LOAD_ASYNC_TO_LDS_B128 when available.
// ---------------------------------------------------------------------------

#define BS 4096
#define DIM 1024
#define NH 4
#define NE 8
#define HD 256
#define CD 1024

typedef __attribute__((ext_vector_type(16))) __bf16 v16bf;
typedef __attribute__((ext_vector_type(8)))  float  v8f;
typedef __attribute__((ext_vector_type(8)))  unsigned short v8us;
typedef __attribute__((ext_vector_type(4)))  float  v4f;
typedef __attribute__((ext_vector_type(4)))  unsigned int v4u;
typedef __attribute__((ext_vector_type(2)))  unsigned int v2u;
typedef __attribute__((ext_vector_type(4)))  int v4i;

__device__ __forceinline__ unsigned short f2bf(float f) {
    unsigned int u = __float_as_uint(f);
    u += 0x7FFFu + ((u >> 16) & 1u);           // round-to-nearest-even
    return (unsigned short)(u >> 16);
}
__device__ __forceinline__ float bf2f(unsigned short b) {
    return __uint_as_float(((unsigned int)b) << 16);
}

// ---- async global->LDS staging (CDNA5), with safe fallback ----------------
#if __has_builtin(__builtin_amdgcn_global_load_async_to_lds_b128)
#define HAVE_ASYNC_LDS 1
typedef __attribute__((address_space(1))) v4i gv4i;   // global 128-bit
typedef __attribute__((address_space(3))) v4i lv4i;   // LDS 128-bit
#else
#define HAVE_ASYNC_LDS 0
#endif

__device__ __forceinline__ void copy16_g2l(unsigned short* ldst,
                                           const unsigned short* gsrc) {
#if HAVE_ASYNC_LDS
    __builtin_amdgcn_global_load_async_to_lds_b128(
        (gv4i*)(v4i*)(void*)gsrc, (lv4i*)(v4i*)(void*)ldst, 0, 0);
#else
    *(v8us*)ldst = *(const v8us*)gsrc;
#endif
}
__device__ __forceinline__ void staging_join() {
#if HAVE_ASYNC_LDS
#if __has_builtin(__builtin_amdgcn_s_wait_asynccnt)
    __builtin_amdgcn_s_wait_asynccnt(0);
#endif
#endif
}

union Frag16 { v8us u[2]; v16bf b; };

// A fragment (16x32 bf16, MxK), LDS row-major [row][k], pitch multiple of 8.
__device__ __forceinline__ v16bf load_a_frag(const unsigned short* lds,
                                             int row0, int pitch, int k0) {
    const int lane = threadIdx.x & 31;
    const int r  = row0 + (lane & 15);
    const int kb = k0 + ((lane & 16) ? 8 : 0);
    const unsigned short* p = lds + r * pitch + kb;
    Frag16 f;
    f.u[0] = *(const v8us*)(p);
    f.u[1] = *(const v8us*)(p + 16);
    return f.b;
}
// B fragment (32x16 bf16, KxN), LDS stored transposed [n][k].
__device__ __forceinline__ v16bf load_b_frag(const unsigned short* lds,
                                             int n0, int pitch, int k0) {
    const int lane = threadIdx.x & 31;
    const int n  = n0 + (lane & 15);
    const int kb = k0 + ((lane & 16) ? 8 : 0);
    const unsigned short* p = lds + n * pitch + kb;
    Frag16 f;
    f.u[0] = *(const v8us*)(p);
    f.u[1] = *(const v8us*)(p + 16);
    return f.b;
}
__device__ __forceinline__ v8f wmma_bf16(v16bf a, v16bf b, v8f c) {
    return __builtin_amdgcn_wmma_f32_16x16x32_bf16(
        false, a, false, b, (short)0, c, false, false);
}

// ---------------------------------------------------------------------------
// Elementwise fp32 -> bf16 (8 elems/thread, 16B stores)
// ---------------------------------------------------------------------------
__global__ __launch_bounds__(256)
void cvt_kernel(const float* __restrict__ in, unsigned short* __restrict__ out,
                int n8) {
    const int i = blockIdx.x * 256 + threadIdx.x;
    if (i >= n8) return;
    const v4f* src = (const v4f*)in + 2 * (size_t)i;
    const v4f a = src[0], b = src[1];
    v4u p;
    p[0] = (unsigned)f2bf(a[0]) | ((unsigned)f2bf(a[1]) << 16);
    p[1] = (unsigned)f2bf(a[2]) | ((unsigned)f2bf(a[3]) << 16);
    p[2] = (unsigned)f2bf(b[0]) | ((unsigned)f2bf(b[1]) << 16);
    p[3] = (unsigned)f2bf(b[2]) | ((unsigned)f2bf(b[3]) << 16);
    ((v4u*)out)[i] = p;
}

// ---------------------------------------------------------------------------
// Batched transpose + convert: in[z][R][C] f32 -> out[z][C][R] bf16
// 32x32 tiles through LDS; coalesced on both sides.
// ---------------------------------------------------------------------------
__global__ __launch_bounds__(256)
void transpose_cvt_kernel(const float* __restrict__ in,
                          unsigned short* __restrict__ out, int R, int C) {
    __shared__ unsigned short tile[32][33];
    const size_t base = (size_t)blockIdx.z * R * C;
    const int r0 = blockIdx.y * 32, c0 = blockIdx.x * 32;
    const int tr = threadIdx.x >> 3;          // 0..31
    const int tc = (threadIdx.x & 7) * 4;     // 0,4,...,28
    const float* src = in + base + (size_t)(r0 + tr) * C + c0 + tc;
    #pragma unroll
    for (int i = 0; i < 4; ++i) tile[tr][tc + i] = f2bf(src[i]);
    __syncthreads();
    unsigned short* dst = out + base + (size_t)(c0 + tr) * R + r0 + tc;
    v2u p;
    p[0] = (unsigned)tile[tc + 0][tr] | ((unsigned)tile[tc + 1][tr] << 16);
    p[1] = (unsigned)tile[tc + 2][tr] | ((unsigned)tile[tc + 3][tr] << 16);
    *(v2u*)dst = p;
}

// ---------------------------------------------------------------------------
// C = A @ W^T   (A: [M,K] bf16 ; W: [N,K] bf16 ; C: [M,N] f32 or bf16)
// 128x128 block, BK=32, 8 waves; wave = 32x64 sub-tile (2x4 fragments)
// ---------------------------------------------------------------------------
template<bool OUT_F32>
__global__ __launch_bounds__(256)
void gemm_xwt_kernel(const unsigned short* __restrict__ A,
                     const unsigned short* __restrict__ W,
                     void* __restrict__ Out, int M, int N, int K) {
    __shared__ unsigned short lsA[128 * 40];   // [128][32] pitch 40
    __shared__ unsigned short lsB[128 * 40];   // [n:128][k:32] pitch 40

    const int tid  = threadIdx.x;
    const int wave = tid >> 5;
    const int m0 = blockIdx.x * 128;
    const int n0 = blockIdx.y * 128;
    const int rowg = wave & 3;
    const int colg = wave >> 2;

    v8f acc[2][4];
    for (int i = 0; i < 2; ++i)
        for (int j = 0; j < 4; ++j)
            for (int r = 0; r < 8; ++r) acc[i][j][r] = 0.0f;

    const int srow = tid >> 1;          // 0..127
    const int skp  = (tid & 1) * 16;    // 0 / 16

    for (int k0 = 0; k0 < K; k0 += 32) {
        {
            const unsigned short* src = A + (size_t)(m0 + srow) * K + k0 + skp;
            unsigned short* dst = lsA + srow * 40 + skp;
            copy16_g2l(dst, src);
            copy16_g2l(dst + 8, src + 8);
        }
        {
            const unsigned short* src = W + (size_t)(n0 + srow) * K + k0 + skp;
            unsigned short* dst = lsB + srow * 40 + skp;
            copy16_g2l(dst, src);
            copy16_g2l(dst + 8, src + 8);
        }
        staging_join();
        __syncthreads();
        #pragma unroll
        for (int i = 0; i < 2; ++i) {
            v16bf a = load_a_frag(lsA, rowg * 32 + i * 16, 40, 0);
            #pragma unroll
            for (int j = 0; j < 4; ++j) {
                v16bf b = load_b_frag(lsB, colg * 64 + j * 16, 40, 0);
                acc[i][j] = wmma_bf16(a, b, acc[i][j]);
            }
        }
        __syncthreads();
    }

    const int lane = tid & 31;
    const int cn   = lane & 15;
    const int mofs = (lane & 16) ? 8 : 0;
    for (int i = 0; i < 2; ++i)
        for (int j = 0; j < 4; ++j) {
            const int gm = m0 + rowg * 32 + i * 16 + mofs;
            const int gn = n0 + colg * 64 + j * 16 + cn;
            if constexpr (OUT_F32) {
                float* O = (float*)Out;
                #pragma unroll
                for (int r = 0; r < 8; ++r)
                    O[(size_t)(gm + r) * N + gn] = acc[i][j][r];
            } else {
                unsigned short* O = (unsigned short*)Out;
                #pragma unroll
                for (int r = 0; r < 8; ++r)
                    O[(size_t)(gm + r) * N + gn] = f2bf(acc[i][j][r]);
            }
        }
}

// ---------------------------------------------------------------------------
// Gating: per (b,h) logits over 8 experts, softmax, top-2, renormalize.
// Writes dense g[b][h][e] (zero except the two selected experts).
// ---------------------------------------------------------------------------
__global__ __launch_bounds__(256)
void gate_kernel(const unsigned short* __restrict__ xh,
                 const float* __restrict__ gate_w, float* __restrict__ g) {
    const int idx = blockIdx.x * 256 + threadIdx.x;   // b*NH + h
    if (idx >= BS * NH) return;
    const int b = idx >> 2, h = idx & 3;
    const unsigned short* xp = xh + (size_t)b * DIM + h * HD;
    const float* gw = gate_w + (size_t)h * HD * NE;

    float lg[NE];
    #pragma unroll
    for (int e = 0; e < NE; ++e) lg[e] = 0.0f;
    for (int d = 0; d < HD; ++d) {
        const float xv = bf2f(xp[d]);
        #pragma unroll
        for (int e = 0; e < NE; ++e) lg[e] += xv * gw[d * NE + e];
    }
    float mx = lg[0];
    #pragma unroll
    for (int e = 1; e < NE; ++e) mx = fmaxf(mx, lg[e]);
    float pe[NE], s = 0.0f;
    #pragma unroll
    for (int e = 0; e < NE; ++e) { pe[e] = __expf(lg[e] - mx); s += pe[e]; }
    const float inv_s = 1.0f / s;
    #pragma unroll
    for (int e = 0; e < NE; ++e) pe[e] *= inv_s;
    int i1 = 0;
    #pragma unroll
    for (int e = 1; e < NE; ++e) if (pe[e] > pe[i1]) i1 = e;
    int i2 = (i1 == 0) ? 1 : 0;
    #pragma unroll
    for (int e = 0; e < NE; ++e) { if (e != i1 && pe[e] > pe[i2]) i2 = e; }
    const float inv2 = 1.0f / (pe[i1] + pe[i2]);
    float* go = g + (size_t)idx * NE;
    #pragma unroll
    for (int e = 0; e < NE; ++e) go[e] = 0.0f;
    go[i1] = pe[i1] * inv2;
    go[i2] = pe[i2] * inv2;
}

// ---------------------------------------------------------------------------
// Fused MoE FFN: per (64-token tile, head). Loops experts; per CD-chunk of 128:
//   up GEMM (K=256) -> relu^2 * gate(row) -> bf16 act in LDS -> down GEMM
// accumulating the 64x256 head output in registers across experts.
// Weights come pre-transposed+bf16: upT[h,e][c][d], dnT[h,e][n][c].
// ---------------------------------------------------------------------------
#define XH_OFF   0                      // [64][264] us  -> 33792 B
#define UP_OFF   (33792 / 2)            // [128][264] us -> 67584 B
#define ACT_OFF  (101376 / 2)           // [64][136] us  -> 17408 B
#define DN_OFF   (118784 / 2)           // [256][136] us -> 69632 B
#define G_OFF    (188416 / 2)           // 64 floats     -> 256 B
#define MOE_LDS_BYTES 188672

__global__ __launch_bounds__(256)
void moe_kernel(const unsigned short* __restrict__ xh,
                const unsigned short* __restrict__ upT,
                const unsigned short* __restrict__ dnT,
                const float* __restrict__ g,
                unsigned short* __restrict__ hout) {
    extern __shared__ unsigned short smem[];
    unsigned short* lds_xh  = smem + XH_OFF;    // pitch 264
    unsigned short* lds_up  = smem + UP_OFF;    // [c][d] pitch 264
    unsigned short* lds_act = smem + ACT_OFF;   // pitch 136
    unsigned short* lds_dn  = smem + DN_OFF;    // [n][c] pitch 136
    float* lds_g = (float*)(smem + G_OFF);      // [64]

    const int tid  = threadIdx.x;
    const int wave = tid >> 5;
    const int lane = tid & 31;
    const int m0 = blockIdx.x * 64;
    const int h  = blockIdx.y;

    // stage xh tile rows m0..m0+63, cols h*HD..h*HD+255
    {
        const int row = tid >> 2;
        const int seg = (tid & 3) * 64;
        const unsigned short* src = xh + (size_t)(m0 + row) * DIM + h * HD + seg;
        unsigned short* dst = lds_xh + row * 264 + seg;
        #pragma unroll
        for (int i = 0; i < 8; ++i) copy16_g2l(dst + i * 8, src + i * 8);
    }
    staging_join();

    const int rowg  = wave >> 1;   // out: 4 groups of 16 rows
    const int colg  = wave & 1;    // out: 2 groups of 128 cols
    const int rowg2 = wave & 3;    // up : 4 groups of 16 rows
    const int colg2 = wave >> 2;   // up : 2 groups of 64 cols
    const int cn    = lane & 15;
    const int mofs  = (lane & 16) ? 8 : 0;

    v8f out_acc[8];
    for (int f = 0; f < 8; ++f)
        for (int r = 0; r < 8; ++r) out_acc[f][r] = 0.0f;

    for (int e = 0; e < NE; ++e) {
        __syncthreads();
        if (tid < 64) lds_g[tid] = g[((size_t)(m0 + tid) * NH + h) * NE + e];
        const unsigned short* upw = upT + (size_t)(h * NE + e) * CD * HD;
        const unsigned short* dnw = dnT + (size_t)(h * NE + e) * HD * CD;

        for (int c0 = 0; c0 < CD; c0 += 128) {
            // ---- stage upT chunk: lds_up[c][d] <- upT[c0+c][d] (vector copy) ----
            {
                const int row = tid >> 1;            // 0..127
                const int seg = (tid & 1) * 128;     // 0 / 128
                const unsigned short* src = upw + (size_t)(c0 + row) * HD + seg;
                unsigned short* dst = lds_up + row * 264 + seg;
                #pragma unroll
                for (int i = 0; i < 16; ++i) copy16_g2l(dst + i * 8, src + i * 8);
            }
            staging_join();
            __syncthreads();
            // ---- up GEMM: 64x128, K=256 ----
            v8f up_acc[4];
            for (int f = 0; f < 4; ++f)
                for (int r = 0; r < 8; ++r) up_acc[f][r] = 0.0f;
            #pragma unroll
            for (int kk = 0; kk < HD; kk += 32) {
                v16bf a = load_a_frag(lds_xh, rowg2 * 16, 264, kk);
                #pragma unroll
                for (int f = 0; f < 4; ++f) {
                    v16bf b = load_b_frag(lds_up, colg2 * 64 + f * 16, 264, kk);
                    up_acc[f] = wmma_bf16(a, b, up_acc[f]);
                }
            }
            // ---- relu^2 * per-row gate -> bf16 act chunk ----
            #pragma unroll
            for (int f = 0; f < 4; ++f) {
                const int col = colg2 * 64 + f * 16 + cn;
                #pragma unroll
                for (int r = 0; r < 8; ++r) {
                    const int row = rowg2 * 16 + mofs + r;
                    float v = up_acc[f][r];
                    v = v > 0.0f ? v : 0.0f;
                    v = v * v * lds_g[row];
                    lds_act[row * 136 + col] = f2bf(v);
                }
            }
            __syncthreads();
            // ---- stage dnT chunk: lds_dn[n][c] <- dnT[n][c0+c] (vector copy) ----
            {
                const int n = tid;                   // 0..255
                const unsigned short* src = dnw + (size_t)n * CD + c0;
                unsigned short* dst = lds_dn + n * 136;
                #pragma unroll
                for (int i = 0; i < 16; ++i) copy16_g2l(dst + i * 8, src + i * 8);
            }
            staging_join();
            __syncthreads();
            // ---- down GEMM: 64x256 out, K=128, accumulate across chunks/experts ----
            #pragma unroll
            for (int kk = 0; kk < 128; kk += 32) {
                v16bf a = load_a_frag(lds_act, rowg * 16, 136, kk);
                #pragma unroll
                for (int f = 0; f < 8; ++f) {
                    v16bf b = load_b_frag(lds_dn, colg * 128 + f * 16, 136, kk);
                    out_acc[f] = wmma_bf16(a, b, out_acc[f]);
                }
            }
            __syncthreads();
        }
    }

    // ---- store head_out tile (bf16) at [b][h*HD + n] ----
    #pragma unroll
    for (int f = 0; f < 8; ++f) {
        const int gn = h * HD + colg * 128 + f * 16 + cn;
        const int gm = m0 + rowg * 16 + mofs;
        #pragma unroll
        for (int r = 0; r < 8; ++r)
            hout[(size_t)(gm + r) * DIM + gn] = f2bf(out_acc[f][r]);
    }
}

// ---------------------------------------------------------------------------
extern "C" void kernel_launch(void* const* d_in, const int* in_sizes, int n_in,
                              void* d_out, int out_size, void* d_ws, size_t ws_size,
                              hipStream_t stream) {
    const float* x      = (const float*)d_in[0];
    const float* w_head = (const float*)d_in[1];
    const float* w_out  = (const float*)d_in[2];
    const float* gate_w = (const float*)d_in[3];
    const float* up_w   = (const float*)d_in[4];
    const float* down_w = (const float*)d_in[5];
    float* out = (float*)d_out;

    // ---- workspace layout (bf16 tensors unless noted) ----
    char* p = (char*)d_ws;
    unsigned short* ws_xbf   = (unsigned short*)p; p += (size_t)BS * DIM * 2;        // 8 MB
    unsigned short* ws_whead = (unsigned short*)p; p += (size_t)DIM * DIM * 2;       // 2 MB
    unsigned short* ws_wout  = (unsigned short*)p; p += (size_t)DIM * DIM * 2;       // 2 MB
    unsigned short* ws_upT   = (unsigned short*)p; p += (size_t)NH * NE * HD * CD * 2; // 16 MB
    unsigned short* ws_dnT   = (unsigned short*)p; p += (size_t)NH * NE * HD * CD * 2; // 16 MB
    unsigned short* ws_xh    = (unsigned short*)p; p += (size_t)BS * DIM * 2;        // 8 MB
    float*          ws_g     = (float*)p;          p += (size_t)BS * NH * NE * 4;    // 0.5 MB
    unsigned short* ws_hout  = (unsigned short*)p;                                   // 8 MB

    (void)hipFuncSetAttribute((const void*)moe_kernel,
                              hipFuncAttributeMaxDynamicSharedMemorySize,
                              MOE_LDS_BYTES);

    // 0) precision/layout prep (bf16 + weight transposes), done every call
    cvt_kernel<<<(BS * DIM / 8 + 255) / 256, 256, 0, stream>>>(x, ws_xbf, BS * DIM / 8);
    cvt_kernel<<<(DIM * DIM / 8 + 255) / 256, 256, 0, stream>>>(w_head, ws_whead, DIM * DIM / 8);
    cvt_kernel<<<(DIM * DIM / 8 + 255) / 256, 256, 0, stream>>>(w_out, ws_wout, DIM * DIM / 8);
    {   // up_w [z][HD][CD] -> upT [z][CD][HD]
        dim3 gT(CD / 32, HD / 32, NH * NE);
        transpose_cvt_kernel<<<gT, 256, 0, stream>>>(up_w, ws_upT, HD, CD);
    }
    {   // down_w [z][CD][HD] -> dnT [z][HD][CD]
        dim3 gT(HD / 32, CD / 32, NH * NE);
        transpose_cvt_kernel<<<gT, 256, 0, stream>>>(down_w, ws_dnT, CD, HD);
    }

    dim3 gridG(BS / 128, DIM / 128);   // 32 x 8
    // 1) xh = x @ w_head^T  (bf16 out)
    gemm_xwt_kernel<false><<<gridG, 256, 0, stream>>>(ws_xbf, ws_whead, (void*)ws_xh,
                                                      BS, DIM, DIM);
    // 2) gating (softmax / top-2 / renorm -> dense gate weights)
    gate_kernel<<<(BS * NH) / 256, 256, 0, stream>>>(ws_xh, gate_w, ws_g);
    // 3) fused MoE FFN
    dim3 gridM(BS / 64, NH);
    moe_kernel<<<gridM, 256, MOE_LDS_BYTES, stream>>>(ws_xh, ws_upT, ws_dnT, ws_g, ws_hout);
    // 4) out = head_out @ w_out^T  (fp32 out)
    gemm_xwt_kernel<true><<<gridG, 256, 0, stream>>>(ws_hout, ws_wout, (void*)out,
                                                     BS, DIM, DIM);
}
